// RBFNetworkBiasedScanAttention_14130442404170
// MI455X (gfx1250) — compile-verified, matrix operand order checked by reading, not run on hardware
//
#include <hip/hip_runtime.h>
#include <hip/hip_bf16.h>

typedef __bf16 v16bf __attribute__((ext_vector_type(16)));
typedef __bf16 v8bf  __attribute__((ext_vector_type(8)));
typedef __bf16 v4bf  __attribute__((ext_vector_type(4)));
typedef float  v8f   __attribute__((ext_vector_type(8)));

#define LOG2E 1.44269504088896340736f

// Problem constants (fixed by reference setup_inputs)
constexpr int B_ = 2, L_ = 8192, H_ = 8, D_ = 64, M_ = 2, F_ = 5;
constexpr int QT = 128;          // q rows per block (8 waves x 16 rows)
constexpr int KT = 32;           // k columns per main-loop iteration
constexpr int NW = QT / 16;      // waves per block
constexpr int KP = 72;           // sK row stride in bf16 (16B-aligned rows, padded)
constexpr int VP = 40;           // sVt row stride in bf16
constexpr int PP = 36;           // sP row stride in f32

static __device__ __forceinline__ v16bf cat8(v8bf lo, v8bf hi) {
    return __builtin_shufflevector(lo, hi, 0,1,2,3,4,5,6,7,8,9,10,11,12,13,14,15);
}

__global__ __launch_bounds__(256)
void rbf_flash_attn(const float* __restrict__ qs, const float* __restrict__ ks,
                    const float* __restrict__ vs, const float* __restrict__ qms,
                    const float* __restrict__ kms, const float* __restrict__ av,
                    const float* __restrict__ bv, float* __restrict__ out)
{
    __shared__ __bf16 sK[KT][KP];       // K tile, row-major [k][d]
    __shared__ __bf16 sVt[D_][VP];      // V tile transposed [d][k]
    __shared__ float  sKM[KT][M_];      // km tile
    __shared__ float  sP[NW][16][PP];   // per-wave P staging (f32)

    const int tid  = threadIdx.x;
    const int lane = tid & 31;
    const int wave = tid >> 5;
    const int b = blockIdx.z, h = blockIdx.y;
    const int q0 = blockIdx.x * QT + wave * 16;

    const int col = lane & 15;          // C-layout column / B-frag column / A-frag row
    const int hi  = lane >> 4;          // half-group select

    // Per-head RBF params: bias = sum_f a_f * exp2(-(b_f^2*log2e) * d2)
    float ah[F_], bsc[F_];
    #pragma unroll
    for (int f = 0; f < F_; ++f) {
        ah[f] = av[h * F_ + f];
        float bf = bv[h * F_ + f];
        bsc[f] = -(bf * bf) * LOG2E;
    }

    // ---- Q fragments (A-matrix, 16x32 bf16 per D-chunk), pre-scaled by 1/sqrt(D) ----
    v16bf qfrag[2];
    {
        const int m = lane & 15;
        const float* qrow = qs + (((size_t)b * L_ + (q0 + m)) * H_ + h) * D_;
        #pragma unroll
        for (int c = 0; c < 2; ++c) {
            const int d0 = c * 32 + hi * 8;
            #pragma unroll
            for (int e = 0; e < 8; ++e) {
                qfrag[c][e]     = (__bf16)(qrow[d0 + e]      * 0.125f);
                qfrag[c][e + 8] = (__bf16)(qrow[d0 + 16 + e] * 0.125f);
            }
        }
    }

    // q-meta for this lane's 8 rows (rows r + 8*hi)
    float qmx[8], qmy[8];
    #pragma unroll
    for (int r = 0; r < 8; ++r) {
        const int qr = q0 + r + 8 * hi;
        qmx[r] = qms[((size_t)b * L_ + qr) * M_ + 0];
        qmy[r] = qms[((size_t)b * L_ + qr) * M_ + 1];
    }

    v8f acc[4];
    #pragma unroll
    for (int n = 0; n < 4; ++n) acc[n] = (v8f){0.f,0.f,0.f,0.f,0.f,0.f,0.f,0.f};
    float mrow[8], srow[8];
    #pragma unroll
    for (int r = 0; r < 8; ++r) { mrow[r] = -INFINITY; srow[r] = 0.f; }

    const int nkt = L_ / KT;
    for (int kt = 0; kt < nkt; ++kt) {
        __syncthreads();   // previous iteration done reading tiles
        // ---- cooperative tile load: 32x64 K (bf16 row-major), 32x64 V (bf16 transposed) ----
        {
            const int kbase = kt * KT;
            #pragma unroll
            for (int it = 0; it < 2; ++it) {
                const int idx = tid * 2 + it;        // 0..511 float4s
                const int kr  = idx >> 4;            // 0..31
                const int d4  = (idx & 15) * 4;      // 0..60
                const size_t go = (((size_t)b * L_ + kbase + kr) * H_ + h) * D_ + d4;
                const float4 kv = *(const float4*)(ks + go);
                v4bf kb; kb[0]=(__bf16)kv.x; kb[1]=(__bf16)kv.y; kb[2]=(__bf16)kv.z; kb[3]=(__bf16)kv.w;
                *(v4bf*)&sK[kr][d4] = kb;
                const float4 vv = *(const float4*)(vs + go);
                sVt[d4 + 0][kr] = (__bf16)vv.x;
                sVt[d4 + 1][kr] = (__bf16)vv.y;
                sVt[d4 + 2][kr] = (__bf16)vv.z;
                sVt[d4 + 3][kr] = (__bf16)vv.w;
            }
            if (tid < KT * M_)
                ((float*)sKM)[tid] = kms[((size_t)b * L_ + kbase) * M_ + tid];
            if (kt + 1 < nkt) {   // prefetch next tile into cache (global_prefetch_b8)
                const size_t gn = (((size_t)b * L_ + kbase + KT + lane) * H_ + h) * D_;
                __builtin_prefetch(ks + gn, 0, 0);
                __builtin_prefetch(vs + gn, 0, 0);
            }
        }
        __syncthreads();

        // ---- S = Q*K^T : two 16x16 f32 tiles (cols 0..15 and 16..31) ----
        v8f s0 = (v8f){0.f,0.f,0.f,0.f,0.f,0.f,0.f,0.f};
        v8f s1 = s0;
        #pragma unroll
        for (int c = 0; c < 2; ++c) {
            const int d0 = c * 32 + hi * 8;
            v16bf b0 = cat8(*(const v8bf*)&sK[col][d0],      *(const v8bf*)&sK[col][d0 + 16]);
            s0 = __builtin_amdgcn_wmma_f32_16x16x32_bf16(false, qfrag[c], false, b0,
                                                         (short)0, s0, false, false);
            v16bf b1 = cat8(*(const v8bf*)&sK[col + 16][d0], *(const v8bf*)&sK[col + 16][d0 + 16]);
            s1 = __builtin_amdgcn_wmma_f32_16x16x32_bf16(false, qfrag[c], false, b1,
                                                         (short)0, s1, false, false);
        }

        // ---- RBF bias ----
        const float kx0 = sKM[col][0],      ky0 = sKM[col][1];
        const float kx1 = sKM[col + 16][0], ky1 = sKM[col + 16][1];
        #pragma unroll
        for (int r = 0; r < 8; ++r) {
            float dx = qmx[r] - kx0, dy = qmy[r] - ky0;
            float d2a = dx * dx + dy * dy;
            dx = qmx[r] - kx1; dy = qmy[r] - ky1;
            float d2b = dx * dx + dy * dy;
            float ba = 0.f, bbx = 0.f;
            #pragma unroll
            for (int f = 0; f < F_; ++f) {
                ba  += ah[f] * __builtin_amdgcn_exp2f(bsc[f] * d2a);
                bbx += ah[f] * __builtin_amdgcn_exp2f(bsc[f] * d2b);
            }
            s0[r] += ba;
            s1[r] += bbx;
        }

        // ---- online softmax update + stage P in LDS ----
        #pragma unroll
        for (int r = 0; r < 8; ++r) {
            float mx = fmaxf(s0[r], s1[r]);
            #pragma unroll
            for (int msk = 8; msk >= 1; msk >>= 1) mx = fmaxf(mx, __shfl_xor(mx, msk, 32));
            const float mnew = fmaxf(mrow[r], mx);
            const float corr = __builtin_amdgcn_exp2f((mrow[r] - mnew) * LOG2E);
            const float p0 = __builtin_amdgcn_exp2f((s0[r] - mnew) * LOG2E);
            const float p1 = __builtin_amdgcn_exp2f((s1[r] - mnew) * LOG2E);
            float rs = p0 + p1;
            #pragma unroll
            for (int msk = 8; msk >= 1; msk >>= 1) rs += __shfl_xor(rs, msk, 32);
            srow[r] = corr * srow[r] + rs;
            mrow[r] = mnew;
            #pragma unroll
            for (int n = 0; n < 4; ++n) acc[n][r] *= corr;
            sP[wave][r + 8 * hi][col]      = p0;
            sP[wave][r + 8 * hi][col + 16] = p1;
        }
        asm volatile("s_wait_dscnt 0" ::: "memory");   // same-wave LDS RAW

        // ---- P (16x32) as A-fragment, then PV: 4 WMMAs over Dv chunks ----
        v16bf pf;
        {
            const float* pr = sP[wave][lane & 15];
            const int k0 = hi * 8;
            #pragma unroll
            for (int e = 0; e < 8; ++e) {
                pf[e]     = (__bf16)pr[k0 + e];
                pf[e + 8] = (__bf16)pr[k0 + 16 + e];
            }
        }
        #pragma unroll
        for (int n = 0; n < 4; ++n) {
            const __bf16* vr = &sVt[n * 16 + col][hi * 8];
            v16bf bf_ = cat8(*(const v8bf*)vr, *(const v8bf*)(vr + 16));
            acc[n] = __builtin_amdgcn_wmma_f32_16x16x32_bf16(false, pf, false, bf_,
                                                             (short)0, acc[n], false, false);
        }
    }

    // ---- epilogue: out[b, q, h, d] = acc / rowsum ----
    #pragma unroll
    for (int r = 0; r < 8; ++r) {
        const float inv = 1.0f / srow[r];
        const int qr = q0 + r + 8 * hi;
        float* orow = out + (((size_t)b * L_ + qr) * H_ + h) * D_;
        #pragma unroll
        for (int n = 0; n < 4; ++n)
            orow[n * 16 + col] = acc[n][r] * inv;
    }
}

extern "C" void kernel_launch(void* const* d_in, const int* in_sizes, int n_in,
                              void* d_out, int out_size, void* d_ws, size_t ws_size,
                              hipStream_t stream) {
    const float* qs  = (const float*)d_in[0];
    const float* ks  = (const float*)d_in[1];
    const float* vs  = (const float*)d_in[2];
    const float* qms = (const float*)d_in[3];
    const float* kms = (const float*)d_in[4];
    const float* a   = (const float*)d_in[5];
    const float* b   = (const float*)d_in[6];
    float* out = (float*)d_out;
    (void)in_sizes; (void)n_in; (void)out_size; (void)d_ws; (void)ws_size;

    dim3 grid(L_ / QT, H_, B_);   // 64 x 8 x 2 blocks
    dim3 block(256);              // 8 waves (wave32)
    rbf_flash_attn<<<grid, block, 0, stream>>>(qs, ks, vs, qms, kms, a, b, out);
}
